// BiMambaRPPGModel_44392781971762
// MI455X (gfx1250) — matrix operand order repaired and checked by reading, not compiled, not run on hardware
//
#include <hip/hip_runtime.h>

// ---------------------------------------------------------------------------
// BiMamba rPPG model for MI455X (gfx1250, wave32, WMMA bf16, async-LDS)
// ---------------------------------------------------------------------------

typedef __bf16 bf16;
typedef __attribute__((ext_vector_type(8)))  bf16  v8bf;
typedef __attribute__((ext_vector_type(16))) bf16  v16bf;
typedef __attribute__((ext_vector_type(8)))  float v8f;

// async-LDS builtin operand types (per hipcc diagnostic: v4i global / LDS ptrs)
typedef int v4i __attribute__((vector_size(4 * sizeof(int))));
typedef __attribute__((address_space(1))) v4i* as1_v4i_p;
typedef __attribute__((address_space(3))) v4i* as3_v4i_p;

#define BATCH   16
#define SEQ     4096
#define DM      128      // d_model
#define DI      256      // d_inner
#define DSTATE  32       // d_state  (== wave32 lanes, one state per lane)
#define DTR     8        // dt_rank
#define KPRE    7        // pre-conv kernel
#define DCONV   8        // mamba depthwise conv kernel
#define NB      6        // blocks
#define MROWS   (BATCH*SEQ)   // 65536 GEMM rows
#define XPN     80            // x_proj N padded 72 -> 80 (5 WMMA col tiles)

#if __has_builtin(__builtin_amdgcn_global_load_async_to_lds_b128)
#define HAS_ASYNC_LDS 1
#else
#define HAS_ASYNC_LDS 0
#endif

// ---------------------------------------------------------------------------
// Wave reduction helper (wave32)
// ---------------------------------------------------------------------------
__device__ __forceinline__ float wave_sum(float v) {
#pragma unroll
    for (int o = 16; o; o >>= 1) v += __shfl_xor(v, o, 32);
    return v;
}

// ---------------------------------------------------------------------------
// bf16 WMMA GEMM:  C[M,N] = A[M,K] * W[N,K]^T
//   A row-major bf16 (stride K), W row-major bf16 (N x K  == B^T, exactly the
//   einsum 'ble,ne->bln' layout of the model weights).
//   One wave -> 16x16 D tile; MREP=4 row tiles per wave share the B fragment.
//   W tile (16 cols x K) staged in LDS once per block (async engine when
//   available: GLOBAL_LOAD_ASYNC_TO_LDS_B128 + s_wait_asynccnt).
//   Output: fp32 (Cf) or bf16 (Cb).
// ---------------------------------------------------------------------------
#define GWAVES 8
#define MREP   4

__global__ __launch_bounds__(256) void gemm_bf16_wmma(
    const bf16* __restrict__ A, const bf16* __restrict__ W,
    float* __restrict__ Cf, bf16* __restrict__ Cb, int N, int K)
{
    __shared__ bf16 wlds[16 * 256];           // up to K=256

    const int tid  = threadIdx.x;
    const int wave = tid >> 5;
    const int lane = tid & 31;
    const int half = lane >> 4;               // 0: lanes 0-15, 1: lanes 16-31
    const int r16  = lane & 15;
    const int col0 = blockIdx.y * 16;

    // ---- stage W tile [16][K] into LDS (16B vectors, K multiple of 8) ----
#if HAS_ASYNC_LDS
    for (int t = tid * 8; t < 16 * K; t += 256 * 8) {
        int r = t / K, k = t % K;
        __builtin_amdgcn_global_load_async_to_lds_b128(
            (as1_v4i_p)(W + (size_t)(col0 + r) * K + k),
            (as3_v4i_p)(wlds + t),
            0, 0);
    }
#if __has_builtin(__builtin_amdgcn_s_wait_asynccnt)
    __builtin_amdgcn_s_wait_asynccnt(0);
#else
    asm volatile("s_wait_asynccnt 0x0" ::: "memory");
#endif
    __syncthreads();
#else
    for (int t = tid * 8; t < 16 * K; t += 256 * 8) {
        int r = t / K, k = t % K;
        *(v8bf*)(wlds + t) = *(const v8bf*)(W + (size_t)(col0 + r) * K + k);
    }
    __syncthreads();
#endif

    const int m0 = (blockIdx.x * GWAVES + wave) * (MREP * 16);

    v8f acc[MREP] = {};

    for (int k0 = 0; k0 < K; k0 += 32) {
        // B fragment: col = r16, K-halves by lane group, contiguous 16 from LDS
        union { v16bf v; v8bf h[2]; } bfr;
        const bf16* bp = wlds + r16 * K + k0 + half * 16;
        bfr.h[0] = *(const v8bf*)(bp);
        bfr.h[1] = *(const v8bf*)(bp + 8);

        // issue all A-fragment loads first (one clause), then the WMMA chain
        union { v16bf v; v8bf h[2]; } afr[MREP];
#pragma unroll
        for (int mi = 0; mi < MREP; ++mi) {
            const int row = m0 + mi * 16 + r16;
            const bf16* ap = A + (size_t)row * K + k0 + half * 16;
            afr[mi].h[0] = *(const v8bf*)(ap);        // K = k0 + half*16 .. +7
            afr[mi].h[1] = *(const v8bf*)(ap + 8);    // K = k0+half*16+8 .. +7
        }
#pragma unroll
        for (int mi = 0; mi < MREP; ++mi)
            acc[mi] = __builtin_amdgcn_wmma_f32_16x16x32_bf16(
                false, afr[mi].v, false, bfr.v, (short)0, acc[mi], false, false);
    }

    // ---- store: VGPR i -> row m0+mi*16 + half*8 + i, col = col0 + r16 ----
    if (Cf) {
#pragma unroll
        for (int mi = 0; mi < MREP; ++mi)
#pragma unroll
            for (int i = 0; i < 8; ++i)
                Cf[(size_t)(m0 + mi * 16 + half * 8 + i) * N + col0 + r16] = acc[mi][i];
    } else {
#pragma unroll
        for (int mi = 0; mi < MREP; ++mi)
#pragma unroll
            for (int i = 0; i < 8; ++i)
                Cb[(size_t)(m0 + mi * 16 + half * 8 + i) * N + col0 + r16] = (bf16)acc[mi][i];
    }
}

// ---------------------------------------------------------------------------
// Weight conversion kernels (fp32 -> bf16, once per launch)
// ---------------------------------------------------------------------------
__global__ void cast_w_bf16(const float* __restrict__ s, bf16* __restrict__ d, int n) {
    int i = blockIdx.x * blockDim.x + threadIdx.x;
    if (i < n) d[i] = (bf16)s[i];
}

// x_proj: [12][72][256] -> zero-padded [12][80][256]
__global__ void cast_xp_pad(const float* __restrict__ s, bf16* __restrict__ d) {
    int i = blockIdx.x * blockDim.x + threadIdx.x;
    if (i >= 12 * XPN * DI) return;
    int k = i % DI, row = (i / DI) % XPN, p = i / (DI * XPN);
    d[i] = (row < 72) ? (bf16)s[((size_t)p * 72 + row) * DI + k] : (bf16)0.0f;
}

// ---------------------------------------------------------------------------
// Pre-conv (K=7, same pad) + exact GELU:  x(B,L,1) -> h(B,L,128)
// ---------------------------------------------------------------------------
__global__ void preconv_gelu(const float* __restrict__ x, const float* __restrict__ w,
                             const float* __restrict__ b, float* __restrict__ h)
{
    int idx = blockIdx.x * blockDim.x + threadIdx.x;   // B*L*DM exact
    int c = idx % DM, l = (idx / DM) % SEQ, bb = idx / (DM * SEQ);
    float a = b[c];
#pragma unroll
    for (int k = 0; k < KPRE; ++k) {
        int ll = l - KPRE / 2 + k;
        if (ll >= 0 && ll < SEQ) a += x[(size_t)bb * SEQ + ll] * w[c * KPRE + k];
    }
    h[idx] = 0.5f * a * (1.0f + erff(a * 0.70710678118654752f));
}

// ---------------------------------------------------------------------------
// h (fp32) -> hbf (bf16), optionally sequence-reversed (backward direction)
// ---------------------------------------------------------------------------
__global__ void cast_rows(const float* __restrict__ s, bf16* __restrict__ d, int rev)
{
    int idx = blockIdx.x * blockDim.x + threadIdx.x;   // B*L*DM exact
    int c = idx % DM, l = (idx / DM) % SEQ, bb = idx / (DM * SEQ);
    int sl = rev ? (SEQ - 1 - l) : l;
    d[idx] = (bf16)s[((size_t)bb * SEQ + sl) * DM + c];
}

// ---------------------------------------------------------------------------
// Causal depthwise conv (K=8) + SiLU: u fp32 (B,L,256) -> ubf bf16
// ---------------------------------------------------------------------------
__global__ void conv_silu(const float* __restrict__ u, const float* __restrict__ cw,
                          const float* __restrict__ cb, bf16* __restrict__ ub)
{
    int idx = blockIdx.x * blockDim.x + threadIdx.x;   // B*L*DI exact
    int d = idx % DI, l = (idx / DI) % SEQ, bb = idx / (DI * SEQ);
    float a = cb[d];
#pragma unroll
    for (int k = 0; k < DCONV; ++k) {
        int ll = l - (DCONV - 1) + k;
        if (ll >= 0) a += u[((size_t)bb * SEQ + ll) * DI + d] * cw[d * DCONV + k];
    }
    a = a / (1.0f + __expf(-a));
    ub[idx] = (bf16)a;
}

// ---------------------------------------------------------------------------
// dt = softplus(dbc[:, :8] @ dtw^T + dtb)   (K=8: plain VALU, not worth WMMA)
// ---------------------------------------------------------------------------
__global__ void dt_softplus(const float* __restrict__ dbc, const float* __restrict__ dtw,
                            const float* __restrict__ dtb, float* __restrict__ dt)
{
    int idx = blockIdx.x * blockDim.x + threadIdx.x;   // B*L*DI exact
    int d = idx % DI;
    size_t base = (size_t)(idx / DI);
    float a = dtb[d];
#pragma unroll
    for (int r = 0; r < DTR; ++r) a += dbc[base * XPN + r] * dtw[d * DTR + r];
    dt[idx] = (a > 20.0f) ? a : log1pf(__expf(a));
}

// ---------------------------------------------------------------------------
// Selective scan: one wave32 per (batch, channel); lane n holds state n.
//   h[n] = h[n]*exp(dt*A[n]) + dt*u*B[n];  y = sum_n h[n]*C[n] + u*D
//   gated with silu(z); writes bf16 y for the out_proj GEMM.
// ---------------------------------------------------------------------------
__global__ void scan_kernel(const float* __restrict__ dt, const bf16* __restrict__ ub,
                            const float* __restrict__ dbc, const float* __restrict__ alog,
                            const float* __restrict__ dpar, const bf16* __restrict__ zb,
                            bf16* __restrict__ yb)
{
    int tid = blockIdx.x * blockDim.x + threadIdx.x;
    int wid = tid >> 5, lane = tid & 31;
    int bb = wid / DI, d = wid % DI;

    const float A  = -__expf(alog[d * DSTATE + lane]);
    const float Dp = dpar[d];
    float hs = 0.0f;
    const size_t rowbase = (size_t)bb * SEQ;

    for (int l = 0; l < SEQ; ++l) {
        const size_t base = rowbase + l;
        const float dtv = dt[base * DI + d];           // broadcast load
        const float uv  = (float)ub[base * DI + d];    // broadcast load
        const float Bv  = dbc[base * XPN + DTR + lane];
        const float Cv  = dbc[base * XPN + DTR + DSTATE + lane];

        hs = hs * __expf(dtv * A) + (dtv * uv) * Bv;
        float p = wave_sum(hs * Cv);
        if (lane == 0) {
            float y = p + uv * Dp;
            float z = (float)zb[base * DI + d];
            y *= z / (1.0f + __expf(-z));
            yb[base * DI + d] = (bf16)y;
        }
    }
}

// ---------------------------------------------------------------------------
// Accumulate direction output into acc (with reversal for backward dir)
// ---------------------------------------------------------------------------
__global__ void accum_dir(const float* __restrict__ dbuf, float* __restrict__ acc,
                          int rev, int add)
{
    int idx = blockIdx.x * blockDim.x + threadIdx.x;   // B*L*DM exact
    int c = idx % DM, l = (idx / DM) % SEQ, bb = idx / (DM * SEQ);
    int sl = rev ? (SEQ - 1 - l) : l;
    float v = dbuf[((size_t)bb * SEQ + sl) * DM + c];
    acc[idx] = add ? (acc[idx] + v) : v;
}

// ---------------------------------------------------------------------------
// h = LayerNorm(0.5*acc + h) * g + b   — one wave per row (128 ch, 4/lane)
// ---------------------------------------------------------------------------
__global__ void ln_residual(const float* __restrict__ acc, float* __restrict__ h,
                            const float* __restrict__ g, const float* __restrict__ bta)
{
    int tid = blockIdx.x * blockDim.x + threadIdx.x;
    int row = tid >> 5, lane = tid & 31;
    size_t base = (size_t)row * DM;
    float v[4], s = 0.0f;
#pragma unroll
    for (int j = 0; j < 4; ++j) {
        int c = j * 32 + lane;
        v[j] = 0.5f * acc[base + c] + h[base + c];
        s += v[j];
    }
    float mu = wave_sum(s) * (1.0f / DM);
    float vs = 0.0f;
#pragma unroll
    for (int j = 0; j < 4; ++j) { float t = v[j] - mu; vs += t * t; }
    float inv = rsqrtf(wave_sum(vs) * (1.0f / DM) + 1e-5f);
#pragma unroll
    for (int j = 0; j < 4; ++j) {
        int c = j * 32 + lane;
        h[base + c] = (v[j] - mu) * inv * g[c] + bta[c];
    }
}

// ---------------------------------------------------------------------------
// Final projection 128 -> 1 : one wave per row
// ---------------------------------------------------------------------------
__global__ void final_proj(const float* __restrict__ h, const float* __restrict__ pw,
                           const float* __restrict__ pb, float* __restrict__ out)
{
    int tid = blockIdx.x * blockDim.x + threadIdx.x;
    int row = tid >> 5, lane = tid & 31;
    float s = 0.0f;
#pragma unroll
    for (int j = 0; j < 4; ++j) {
        int c = j * 32 + lane;
        s += h[(size_t)row * DM + c] * pw[c];
    }
    s = wave_sum(s);
    if (lane == 0) out[row] = s + pb[0];
}

// ---------------------------------------------------------------------------
// Orchestration
// ---------------------------------------------------------------------------
extern "C" void kernel_launch(void* const* d_in, const int* in_sizes, int n_in,
                              void* d_out, int out_size, void* d_ws, size_t ws_size,
                              hipStream_t stream)
{
    const float* x    = (const float*)d_in[0];
    const float* pcw  = (const float*)d_in[1];
    const float* pcb  = (const float*)d_in[2];
    const float* inw  = (const float*)d_in[3];   // (6,2,512,128)
    const float* cw   = (const float*)d_in[4];   // (6,2,256,1,8)
    const float* cbp  = (const float*)d_in[5];   // (6,2,256)
    const float* xpw  = (const float*)d_in[6];   // (6,2,72,256)
    const float* dtw  = (const float*)d_in[7];   // (6,2,256,8)
    const float* dtb  = (const float*)d_in[8];   // (6,2,256)
    const float* alog = (const float*)d_in[9];   // (6,2,256,32)
    const float* dpar = (const float*)d_in[10];  // (6,2,256)
    const float* outw = (const float*)d_in[11];  // (6,2,128,256)
    const float* lng  = (const float*)d_in[12];  // (6,128)
    const float* lnb  = (const float*)d_in[13];  // (6,128)
    const float* pw   = (const float*)d_in[14];  // (1,128)
    const float* pb   = (const float*)d_in[15];  // (1,)
    float* out = (float*)d_out;

    // ---- workspace carve-up (256B aligned) ----
    char* ws = (char*)d_ws;
    size_t off = 0;
    auto take = [&](size_t bytes) {
        char* p = ws + off;
        off = (off + bytes + 255) & ~(size_t)255;
        return p;
    };
    float* h     = (float*)take((size_t)MROWS * DM * 4);   // residual stream
    bf16*  hbf   = (bf16*) take((size_t)MROWS * DM * 2);   // GEMM1 input
    float* u_dt  = (float*)take((size_t)MROWS * DI * 4);   // u fp32, reused as dt fp32
    bf16*  zbf   = (bf16*) take((size_t)MROWS * DI * 2);   // gate branch
    bf16*  ubf   = (bf16*) take((size_t)MROWS * DI * 2);   // conv+silu output
    float* dbc   = (float*)take((size_t)MROWS * XPN * 4);  // x_proj output (stride 80)
    bf16*  ybf   = (bf16*) take((size_t)MROWS * DI * 2);   // gated scan output
    float* dbuf  = (float*)take((size_t)MROWS * DM * 4);   // per-direction out_proj
    float* acc   = (float*)take((size_t)MROWS * DM * 4);   // fwd+bwd accumulator
    bf16*  w_in  = (bf16*) take((size_t)NB * 2 * 2 * DI * DM * 2);
    bf16*  w_xp  = (bf16*) take((size_t)NB * 2 * XPN * DI * 2);
    bf16*  w_out = (bf16*) take((size_t)NB * 2 * DM * DI * 2);

    // ---- convert weights to bf16 once ----
    {
        int n1 = NB * 2 * 2 * DI * DM;
        cast_w_bf16<<<(n1 + 255) / 256, 256, 0, stream>>>(inw, w_in, n1);
        int n2 = 12 * XPN * DI;
        cast_xp_pad<<<(n2 + 255) / 256, 256, 0, stream>>>(xpw, w_xp);
        int n3 = NB * 2 * DM * DI;
        cast_w_bf16<<<(n3 + 255) / 256, 256, 0, stream>>>(outw, w_out, n3);
    }

    // ---- pre-conv + GELU ----
    preconv_gelu<<<(MROWS * DM) / 256, 256, 0, stream>>>(x, pcw, pcb, h);

    auto gemm = [&](const bf16* A, const bf16* W, float* Cf, bf16* Cb, int N, int K) {
        dim3 grid(MROWS / (GWAVES * MREP * 16), N / 16);   // 128 x (N/16)
        gemm_bf16_wmma<<<grid, 256, 0, stream>>>(A, W, Cf, Cb, N, K);
    };

    for (int i = 0; i < NB; ++i) {
        for (int s = 0; s < 2; ++s) {
            const size_t p = (size_t)(i * 2 + s);

            // bf16 input (reversed for backward direction)
            cast_rows<<<(MROWS * DM) / 256, 256, 0, stream>>>(h, hbf, s);

            // in_proj split: u (fp32, conv input) and z (bf16, gate)
            gemm(hbf, w_in + p * 2 * DI * DM,            u_dt,    nullptr, DI, DM);
            gemm(hbf, w_in + p * 2 * DI * DM + DI * DM,  nullptr, zbf,     DI, DM);

            // causal depthwise conv + silu
            conv_silu<<<(MROWS * DI) / 256, 256, 0, stream>>>(
                u_dt, cw + p * DI * DCONV, cbp + p * DI, ubf);

            // x_proj -> [dt_raw | B | C] (stride 80, padded)
            gemm(ubf, w_xp + p * XPN * DI, dbc, nullptr, XPN, DI);

            // dt = softplus(dbc[:, :8] @ dtw^T + dtb)  (overwrites u_dt)
            dt_softplus<<<(MROWS * DI) / 256, 256, 0, stream>>>(
                dbc, dtw + p * DI * DTR, dtb + p * DI, u_dt);

            // selective scan + D skip + silu(z) gate -> ybf
            scan_kernel<<<(BATCH * DI * 32) / 256, 256, 0, stream>>>(
                u_dt, ubf, dbc, alog + p * DI * DSTATE, dpar + p * DI, zbf, ybf);

            // out_proj
            gemm(ybf, w_out + p * DM * DI, dbuf, nullptr, DM, DI);

            // accumulate (reverse the backward direction back)
            accum_dir<<<(MROWS * DM) / 256, 256, 0, stream>>>(dbuf, acc, s, s);
        }
        // h = LN(0.5*(f+b) + h)
        ln_residual<<<MROWS / 8, 256, 0, stream>>>(acc, h, lng + i * DM, lnb + i * DM);
    }

    // final 128 -> 1 projection
    final_proj<<<MROWS / 8, 256, 0, stream>>>(h, pw, pb, out);
}
